// SearchCNNController_puredarts_14474039787880
// MI455X (gfx1250) — matrix operand order repaired and checked by reading, compiled, not run on hardware
//
#include <hip/hip_runtime.h>

// ---------------------------------------------------------------------------
// DARTS supernet forward for MI455X (gfx1250, wave32).
// All 1x1 (pointwise) convolutions -> v_wmma_f32_16x16x32_bf16 GEMMs with a
// software-pipelined GLOBAL_LOAD_ASYNC_TO_LDS_B64 (ASYNCcnt) double buffer.
// Depthwise convs / pools / BN / softmax / head -> VALU kernels (pow2 shifts,
// 3D grids).
// ---------------------------------------------------------------------------

typedef __attribute__((ext_vector_type(16))) __bf16 v16bf;
typedef __attribute__((ext_vector_type(8)))  float  v8f;

__device__ __forceinline__ unsigned short f2bf(float f) {
  unsigned int u = __float_as_uint(f);
  unsigned int r = u + 0x7fffu + ((u >> 16) & 1u);   // round-to-nearest-even
  return (unsigned short)(r >> 16);
}

__device__ __forceinline__ void wait_async_le0() {
#if __has_builtin(__builtin_amdgcn_s_wait_asynccnt)
  __builtin_amdgcn_s_wait_asynccnt(0);
#else
  asm volatile("s_wait_asynccnt 0" ::: "memory");
#endif
}
__device__ __forceinline__ void wait_async_le8() {
#if __has_builtin(__builtin_amdgcn_s_wait_asynccnt)
  __builtin_amdgcn_s_wait_asynccnt(8);
#else
  asm volatile("s_wait_asynccnt 8" ::: "memory");
#endif
}

// Async DMA: 8 bytes global -> LDS per lane (VDST = LDS byte address VGPR).
__device__ __forceinline__ void async_g2l_b64(unsigned lds_addr, const float* g) {
  asm volatile("global_load_async_to_lds_b64 %0, %1, off"
               :: "v"(lds_addr), "v"((unsigned long long)(size_t)g)
               : "memory");
}

// ---------------------------------------------------------------------------
// Pointwise conv as WMMA GEMM:
//   Y[n, Ycoff+co, ho, wo] = sum_ci W[co,ci] * act(X[n, Xcoff+ci, ho*s+off, wo*s+off])
// Block = 128 threads = 4 waves; each wave owns one 16-pixel tile; grid.y tiles Cout.
// Full-K weight tile staged once (single barrier); B tiles double-buffered and
// pipelined through the async-DMA engine on the stride-1 fast path.
// nshift = log2(Hout*Wout), wshift = log2(Wout)  (all spatial sizes are pow2).
// ---------------------------------------------------------------------------
__global__ __launch_bounds__(128)
void k_pwconv(const float* __restrict__ X, const float* __restrict__ Wt,
              float* __restrict__ Y, int Cin, int XCtot, int Xcoff,
              int Hin, int Win, int Wout, int nshift, int wshift,
              int stride, int off, int relu, int YCtot, int Ycoff)
{
  __shared__ unsigned short As[16][258];   // full-K bf16 weights (conflict-free rows)
  __shared__ float Bsf[2][4][32][18];      // double-buffered fp32 B tile per wave

  const int wave = threadIdx.x >> 5;
  const int lane = threadIdx.x & 31;
  const int HWout = 1 << nshift;
  const int HWin  = Hin * Win;
  const int co0 = blockIdx.y << 4;
  const int p0  = (blockIdx.x * 4 + wave) << 4;   // 16-pixel tile (never crosses n)
  const int n   = p0 >> nshift;
  const int hw0 = p0 & (HWout - 1);
  const bool fast = (stride == 1) && (off == 0);  // contiguous pixel columns
  const float* xnb = X + (size_t)n * XCtot * HWin + (size_t)Xcoff * HWin;
  const int g = lane >> 4, r = lane & 15;
  const int Cpad = (Cin + 31) & ~31;

  __builtin_prefetch(Wt + (size_t)co0 * Cin, 0, 0);   // global_prefetch_b8

  // ---- stage the FULL A tile once (zero-pad K to Cpad) ----
  for (int t = threadIdx.x; t < 16 * Cpad; t += 128) {
    int m = t / Cpad, k = t - m * Cpad;
    float v = (k < Cin) ? Wt[(co0 + m) * Cin + k] : 0.0f;
    As[m][k] = f2bf(v);
  }

  // ---- async prologue: DMA chunk 0 into buffer 0 (2 floats/lane/op) ----
  const int rdma = lane >> 3;           // 0..3 (4 channels per op)
  const int px0  = (lane & 7) << 1;     // 0,2,...,14
  if (fast) {
    for (int q = 0; q < 8; ++q) {
      int rr = (q << 2) + rdma;
      int ci = (rr < Cin) ? rr : Cin - 1;       // clamp: junk zeroed at frag build
      async_g2l_b64((unsigned)(size_t)&Bsf[0][wave][rr][px0],
                    xnb + (size_t)ci * HWin + hw0 + px0);
    }
  }
  __syncthreads();   // only barrier in the kernel (A tile visibility)

  v8f acc;
  #pragma unroll
  for (int i = 0; i < 8; ++i) acc[i] = 0.0f;

  int buf = 0;
  for (int klo = 0; klo < Cin; klo += 32, buf ^= 1) {
    const bool last = (klo + 32 >= Cin);
    if (fast) {
      if (!last) {
        // issue next chunk's DMA into the other buffer, then wait for current
        int knext = klo + 32;
        for (int q = 0; q < 8; ++q) {
          int rr = (q << 2) + rdma;
          int ci = knext + rr; if (ci >= Cin) ci = Cin - 1;
          async_g2l_b64((unsigned)(size_t)&Bsf[buf ^ 1][wave][rr][px0],
                        xnb + (size_t)ci * HWin + hw0 + px0);
        }
        wait_async_le8();   // in-order completion => current chunk landed
      } else {
        wait_async_le0();
      }
    } else {
      // generic strided path (fact_reduce): branch-free, pow2 shifts
      int ci = klo + lane; if (ci >= Cin) ci = Cin - 1;
      const float* xb = xnb + (size_t)ci * HWin;
      for (int j = 0; j < 16; ++j) {
        int hw = hw0 + j;
        int ho = hw >> wshift, wo = hw & (Wout - 1);
        Bsf[buf][wave][lane][j] = xb[(ho * stride + off) * Win + (wo * stride + off)];
      }
    }

    // ---- build fragments (packed 32-bit LDS reads) ----
    union { v16bf v; unsigned int u[8]; } af, bfrag;
    const unsigned int* arow = (const unsigned int*)&As[r][0];
    #pragma unroll
    for (int t = 0; t < 8; ++t) {
      // A layout: element i -> K = klo + i + 16g; pairs K-consecutive.
      af.u[t] = arow[(klo >> 1) + t + 8 * g];
    }
    #pragma unroll
    for (int t = 0; t < 8; ++t) {
      int row0 = 2 * t + 16 * g;        // B layout: element i -> K = i + 16g, N = r
      float v0 = Bsf[buf][wave][row0][r];
      float v1 = Bsf[buf][wave][row0 + 1][r];
      if (klo + row0     >= Cin) v0 = 0.0f;   // kill clamped/stale values in pad chunk
      if (klo + row0 + 1 >= Cin) v1 = 0.0f;
      if (relu) { v0 = fmaxf(v0, 0.0f); v1 = fmaxf(v1, 0.0f); }
      bfrag.u[t] = (unsigned)f2bf(v0) | ((unsigned)f2bf(v1) << 16);
    }

    acc = __builtin_amdgcn_wmma_f32_16x16x32_bf16(
        false, af.v, false, bfrag.v, (short)0, acc, false, false);
  }

  float* yb = Y + (size_t)n * YCtot * HWout + (size_t)(Ycoff + co0) * HWout + hw0 + r;
  #pragma unroll
  for (int i = 0; i < 8; ++i)
    yb[(size_t)(i + 8 * g) * HWout] = acc[i];   // D: row = i + 8g, col = r
}

// ---------------------------------------------------------------------------
// Depthwise conv (ReLU on input). 3D grid: (HWout/bd, C, N); exact, no guards.
// ---------------------------------------------------------------------------
__global__ void k_dwconv(const float* __restrict__ X, const float* __restrict__ Wd,
                         float* __restrict__ Y, int C, int XCtot, int Xcoff,
                         int Hin, int Win, int Wout, int wshift, int HWout,
                         int K, int stride, int pad, int dil)
{
  int hw = blockIdx.x * blockDim.x + threadIdx.x;
  int c = blockIdx.y, n = blockIdx.z;
  int ho = hw >> wshift, wo = hw & (Wout - 1);
  const float* xb = X + (size_t)n * XCtot * Hin * Win + (size_t)(Xcoff + c) * Hin * Win;
  const float* wb = Wd + c * K * K;
  float s = 0.0f;
  for (int kh = 0; kh < K; ++kh) {
    int hi = ho * stride - pad + kh * dil;
    if (hi < 0 || hi >= Hin) continue;
    for (int kw = 0; kw < K; ++kw) {
      int wi = wo * stride - pad + kw * dil;
      if (wi < 0 || wi >= Win) continue;
      s += fmaxf(xb[hi * Win + wi], 0.0f) * wb[kh * K + kw];
    }
  }
  Y[((size_t)n * C + c) * HWout + hw] = s;
}

// 3x3 pool, pad 1 (max with -inf pad / avg with valid-count divisor)
__global__ void k_pool(const float* __restrict__ X, float* __restrict__ Y,
                       int C, int XCtot, int Xcoff, int Hin, int Win,
                       int Wout, int wshift, int HWout, int stride, int ismax)
{
  int hw = blockIdx.x * blockDim.x + threadIdx.x;
  int c = blockIdx.y, n = blockIdx.z;
  int ho = hw >> wshift, wo = hw & (Wout - 1);
  const float* xb = X + (size_t)n * XCtot * Hin * Win + (size_t)(Xcoff + c) * Hin * Win;
  float m = -3.402823466e38f, s = 0.0f; int cnt = 0;
  for (int kh = 0; kh < 3; ++kh) {
    int hi = ho * stride - 1 + kh;
    if (hi < 0 || hi >= Hin) continue;
    for (int kw = 0; kw < 3; ++kw) {
      int wi = wo * stride - 1 + kw;
      if (wi < 0 || wi >= Win) continue;
      float v = xb[hi * Win + wi];
      m = fmaxf(m, v); s += v; cnt++;
    }
  }
  Y[((size_t)n * C + c) * HWout + hw] = ismax ? m : s / (float)cnt;
}

// Batch-norm statistics: one block per channel, reduce over N*HW (pow2 HW).
__global__ __launch_bounds__(256)
void k_bnstats(const float* __restrict__ X, float* __restrict__ meanv,
               float* __restrict__ varv, int C, int HW, int hwshift)
{
  __shared__ float ss[256], sq[256];
  int c = blockIdx.x, tid = threadIdx.x;
  int M = 32 * HW;
  float s = 0.0f, q = 0.0f;
  for (int i = tid; i < M; i += 256) {
    int n = i >> hwshift, hw = i & (HW - 1);
    float v = X[(size_t)n * C * HW + (size_t)c * HW + hw];
    s += v; q += v * v;
  }
  ss[tid] = s; sq[tid] = q; __syncthreads();
  for (int st = 128; st > 0; st >>= 1) {
    if (tid < st) { ss[tid] += ss[tid + st]; sq[tid] += sq[tid + st]; }
    __syncthreads();
  }
  if (tid == 0) {
    float mu = ss[0] / (float)M;
    meanv[c] = mu;
    varv[c]  = sq[0] / (float)M - mu * mu;
  }
}

// Normalize (+ optional affine), write or weighted-accumulate. Grid (HW/bd, C, N).
__global__ void k_bnapply(const float* __restrict__ X, float* __restrict__ OUT,
                          const float* __restrict__ mean, const float* __restrict__ var,
                          const float* __restrict__ gamma, const float* __restrict__ beta,
                          const float* __restrict__ wptr, int widx,
                          int C, int HW, int OCtot, int Ocoff)
{
  int hw = blockIdx.x * blockDim.x + threadIdx.x;
  int c = blockIdx.y, n = blockIdx.z;
  float y = (X[((size_t)n * C + c) * HW + hw] - mean[c]) * rsqrtf(var[c] + 1e-5f);
  if (gamma) y = y * gamma[c] + beta[c];
  size_t o = (size_t)n * OCtot * HW + (size_t)(Ocoff + c) * HW + hw;
  if (wptr) OUT[o] += wptr[widx] * y; else OUT[o] = y;
}

// out += w * x  (identity skip op). Grid (HW/bd, C, N).
__global__ void k_scale_acc(const float* __restrict__ X, float* __restrict__ OUT,
                            const float* __restrict__ wptr, int widx,
                            int C, int XCtot, int Xcoff, int HW,
                            int OCtot, int Ocoff)
{
  int hw = blockIdx.x * blockDim.x + threadIdx.x;
  int c = blockIdx.y, n = blockIdx.z;
  float v = X[(size_t)n * XCtot * HW + (size_t)(Xcoff + c) * HW + hw];
  OUT[(size_t)n * OCtot * HW + (size_t)(Ocoff + c) * HW + hw] += wptr[widx] * v;
}

__global__ void k_zero(float* __restrict__ p, int total) {
  int idx = blockIdx.x * blockDim.x + threadIdx.x;
  if (idx < total) p[idx] = 0.0f;
}

__global__ void k_softmax(const float* __restrict__ a, float* __restrict__ w,
                          int rows, int cols)
{
  int r = threadIdx.x;
  if (r >= rows) return;
  const float* ar = a + r * cols;
  float* wr = w + r * cols;
  float m = ar[0];
  for (int i = 1; i < cols; ++i) m = fmaxf(m, ar[i]);
  float s = 0.0f;
  for (int i = 0; i < cols; ++i) { float e = expf(ar[i] - m); wr[i] = e; s += e; }
  for (int i = 0; i < cols; ++i) wr[i] /= s;
}

// Stem: 3x3 conv, Cin=3 -> 48, pad 1, 32x32. Grid (1024/256, 48, 32).
__global__ void k_stemconv(const float* __restrict__ X, const float* __restrict__ Wt,
                           float* __restrict__ Y)
{
  int hw = blockIdx.x * blockDim.x + threadIdx.x;
  int co = blockIdx.y, n = blockIdx.z;
  int ho = hw >> 5, wo = hw & 31;
  float s = 0.0f;
  for (int ci = 0; ci < 3; ++ci)
    for (int kh = 0; kh < 3; ++kh) {
      int hi = ho - 1 + kh;
      if (hi < 0 || hi >= 32) continue;
      for (int kw = 0; kw < 3; ++kw) {
        int wi = wo - 1 + kw;
        if (wi < 0 || wi >= 32) continue;
        s += X[((size_t)n * 3 + ci) * 1024 + hi * 32 + wi] *
             Wt[co * 27 + ci * 9 + kh * 3 + kw];
      }
    }
  Y[((size_t)n * 48 + co) * 1024 + hw] = s;
}

__global__ void k_gap(const float* __restrict__ X, float* __restrict__ F,
                      int HW, int total)   // C fixed at 256
{
  int idx = blockIdx.x * blockDim.x + threadIdx.x;
  if (idx >= total) return;
  int c = idx & 255, n = idx >> 8;
  float s = 0.0f;
  const float* xb = X + ((size_t)n * 256 + c) * HW;
  for (int i = 0; i < HW; ++i) s += xb[i];
  F[idx] = s / (float)HW;
}

__global__ void k_linear(const float* __restrict__ F, const float* __restrict__ LW,
                         const float* __restrict__ LB, float* __restrict__ OUT,
                         int C, int total)
{
  int idx = blockIdx.x * blockDim.x + threadIdx.x;
  if (idx >= total) return;
  int k = idx % 10, n = idx / 10;
  float s = LB[k];
  for (int c = 0; c < C; ++c) s += F[n * C + c] * LW[k * C + c];
  OUT[idx] = s;
}

// ---------------------------------------------------------------------------
// Host-side orchestration
// ---------------------------------------------------------------------------
namespace {
struct SepP { const float *dw1, *dw2, *pw1, *pw2; };
struct DilP { const float *dw, *pw; };
struct MixP { DilP d3, d5; SepP s3, s5; const float *sk1, *sk2; };
struct CellP { MixP dag[4][5]; const float *p0w, *p0w1, *p0w2, *p1w; };
struct Cfg { int Cpp, Cp, C; bool red_p, red; };

inline int ilog2(int v) { int s = 0; while ((1 << s) < v) ++s; return s; }
inline int ebd(int HW)  { return HW < 256 ? HW : 256; }

inline void gemm(hipStream_t s, const float* X, const float* Wt, float* Y,
                 int Cin, int Cout, int XCtot, int Xcoff, int Hin, int Hout,
                 int stride, int off, int relu, int YCtot, int Ycoff) {
  dim3 g((32 * Hout * Hout) / 64, Cout / 16);
  k_pwconv<<<g, 128, 0, s>>>(X, Wt, Y, Cin, XCtot, Xcoff, Hin, Hin,
                             Hout, ilog2(Hout * Hout), ilog2(Hout),
                             stride, off, relu, YCtot, Ycoff);
}
inline void bnstats(hipStream_t s, const float* X, float* mean, float* var,
                    int C, int HW) {
  k_bnstats<<<C, 256, 0, s>>>(X, mean, var, C, HW, ilog2(HW));
}
inline void bnapply(hipStream_t s, const float* X, float* OUT, const float* mean,
                    const float* var, const float* g, const float* b,
                    const float* wp, int wi, int C, int HW, int OC, int Oo) {
  int bd = ebd(HW);
  dim3 gr(HW / bd, C, 32);
  k_bnapply<<<gr, bd, 0, s>>>(X, OUT, mean, var, g, b, wp, wi, C, HW, OC, Oo);
}
inline void dwc(hipStream_t s, const float* X, const float* Wd, float* Y, int C,
                int XCtot, int Xcoff, int Hin, int Hout, int K, int stride,
                int pad, int dil) {
  int HWout = Hout * Hout, bd = ebd(HWout);
  dim3 gr(HWout / bd, C, 32);
  k_dwconv<<<gr, bd, 0, s>>>(X, Wd, Y, C, XCtot, Xcoff, Hin, Hin,
                             Hout, ilog2(Hout), HWout, K, stride, pad, dil);
}
inline void pool(hipStream_t s, const float* X, float* Y, int C, int XCtot,
                 int Xcoff, int Hin, int Hout, int stride, int ismax) {
  int HWout = Hout * Hout, bd = ebd(HWout);
  dim3 gr(HWout / bd, C, 32);
  k_pool<<<gr, bd, 0, s>>>(X, Y, C, XCtot, Xcoff, Hin, Hin,
                           Hout, ilog2(Hout), HWout, stride, ismax);
}
inline void sacc(hipStream_t s, const float* X, float* OUT, const float* wp,
                 int wi, int C, int XCtot, int Xcoff, int HW, int OC, int Oo) {
  int bd = ebd(HW);
  dim3 gr(HW / bd, C, 32);
  k_scale_acc<<<gr, bd, 0, s>>>(X, OUT, wp, wi, C, XCtot, Xcoff, HW, OC, Oo);
}
inline void zero(hipStream_t s, float* p, size_t n) {
  k_zero<<<(int)((n + 255) / 256), 256, 0, s>>>(p, (int)n);
}
} // namespace

extern "C" void kernel_launch(void* const* d_in, const int* in_sizes, int n_in,
                              void* d_out, int out_size, void* d_ws, size_t ws_size,
                              hipStream_t stream) {
  (void)in_sizes; (void)n_in; (void)out_size; (void)ws_size;
  const float* x  = (const float*)d_in[0];
  const float* an = (const float*)d_in[1];
  const float* ar = (const float*)d_in[2];

  // --- cell configs (mirrors cell_cfgs()) ---
  Cfg cfg[4];
  {
    int Cpp = 48, Cp = 48, Cc = 16; bool rp = false;
    for (int i = 0; i < 4; ++i) {
      bool rd = (i == 1 || i == 2);
      if (rd) Cc *= 2;
      cfg[i] = {Cpp, Cp, Cc, rp, rd};
      rp = rd; Cpp = Cp; Cp = Cc * 4;
    }
  }

  // --- walk params pytree (sorted dict keys, lists in order) ---
  int pidx = 3;
  auto nx = [&]() { return (const float*)d_in[pidx++]; };
  CellP cp[4];
  for (int c = 0; c < 4; ++c) {
    for (int i = 0; i < 4; ++i)
      for (int j = 0; j < 2 + i; ++j) {
        bool s2 = cfg[c].red && j < 2;
        MixP& m = cp[c].dag[i][j];
        m.d3.dw = nx(); m.d3.pw = nx();
        m.d5.dw = nx(); m.d5.pw = nx();
        m.s3.dw1 = nx(); m.s3.dw2 = nx(); m.s3.pw1 = nx(); m.s3.pw2 = nx();
        m.s5.dw1 = nx(); m.s5.dw2 = nx(); m.s5.pw1 = nx(); m.s5.pw2 = nx();
        if (s2) { m.sk1 = nx(); m.sk2 = nx(); } else { m.sk1 = m.sk2 = nullptr; }
      }
    if (cfg[c].red_p) { cp[c].p0w1 = nx(); cp[c].p0w2 = nx(); cp[c].p0w = nullptr; }
    else             { cp[c].p0w = nx();  cp[c].p0w1 = cp[c].p0w2 = nullptr; }
    cp[c].p1w = nx();
  }
  const float* lin_b  = nx();
  const float* lin_w  = nx();
  const float* stem_b = nx();
  const float* stem_g = nx();
  const float* stem_w = nx();

  // --- workspace bump allocator ---
  float* Wf = (float*)d_ws; size_t wo = 0;
  auto alloc = [&](size_t n) { float* p = Wf + wo; wo += n; return p; };
  float* stem = alloc((size_t)32 * 48 * 1024);
  float* cellout[4];
  cellout[0] = alloc((size_t)32 * 64 * 1024);
  cellout[1] = alloc((size_t)32 * 128 * 256);
  cellout[2] = alloc((size_t)32 * 256 * 64);
  cellout[3] = alloc((size_t)32 * 256 * 64);
  float* s0p = alloc((size_t)32 * 32 * 1024);
  float* s1p = alloc((size_t)32 * 32 * 1024);
  float* tA  = alloc((size_t)32 * 16 * 1024);
  float* tB  = alloc((size_t)32 * 16 * 1024);
  float* mean = alloc(256); float* var = alloc(256);
  float* wn = alloc(14 * 8); float* wr = alloc(14 * 8);
  float* feat = alloc((size_t)32 * 256);

  // --- alpha softmax ---
  k_softmax<<<1, 32, 0, stream>>>(an, wn, 14, 8);
  k_softmax<<<1, 32, 0, stream>>>(ar, wr, 14, 8);

  // --- stem conv + affine BN ---
  {
    dim3 gr(1024 / 256, 48, 32);
    k_stemconv<<<gr, 256, 0, stream>>>(x, stem_w, stem);
    bnstats(stream, stem, mean, var, 48, 1024);
    bnapply(stream, stem, stem, mean, var, stem_g, stem_b, nullptr, 0, 48, 1024, 48, 0);
  }

  static const int EOFF[5] = {0, 2, 5, 9, 14};
  const float* pp = stem; const float* pv = stem;
  int Hpp = 32, Hp = 32;

  for (int c = 0; c < 4; ++c) {
    const Cfg cf = cfg[c]; CellP& P = cp[c];
    const int C = cf.C;
    const int Hn = cf.red ? Hp / 2 : Hp;
    const int HWp = Hp * Hp, HWn = Hn * Hn;
    float* out = cellout[c];
    const int OC4 = 4 * C;

    // pre0 -> s0p (C ch @ Hp)
    if (cf.red_p) {
      int h = C / 2;
      gemm(stream, pp, P.p0w1, s0p, cf.Cpp, h,     cf.Cpp, 0, Hpp, Hp, 2, 0, 1, C, 0);
      gemm(stream, pp, P.p0w2, s0p, cf.Cpp, C - h, cf.Cpp, 0, Hpp, Hp, 2, 1, 1, C, h);
    } else {
      gemm(stream, pp, P.p0w, s0p, cf.Cpp, C, cf.Cpp, 0, Hpp, Hp, 1, 0, 1, C, 0);
    }
    bnstats(stream, s0p, mean, var, C, HWp);
    bnapply(stream, s0p, s0p, mean, var, nullptr, nullptr, nullptr, 0, C, HWp, C, 0);

    // pre1 -> s1p
    gemm(stream, pv, P.p1w, s1p, cf.Cp, C, cf.Cp, 0, Hp, Hp, 1, 0, 1, C, 0);
    bnstats(stream, s1p, mean, var, C, HWp);
    bnapply(stream, s1p, s1p, mean, var, nullptr, nullptr, nullptr, 0, C, HWp, C, 0);

    zero(stream, out, (size_t)32 * OC4 * HWn);
    const float* ws = cf.red ? wr : wn;

    struct St { const float* p; int Ct, co, H; };
    St st[6] = { {s0p, C, 0, Hp}, {s1p, C, 0, Hp},
                 {out, OC4, 0, Hn}, {out, OC4, C, Hn},
                 {out, OC4, 2 * C, Hn}, {out, OC4, 3 * C, Hn} };

    for (int i = 0; i < 4; ++i) {
      int no = i * C;                       // node channel slice in concat output
      for (int j = 0; j < 2 + i; ++j) {
        int sstr = (cf.red && j < 2) ? 2 : 1;
        St in = st[j];
        MixP& m = P.dag[i][j];
        int wb = (EOFF[i] + j) * 8;

        // op0: max pool -> BN -> acc
        pool(stream, in.p, tA, C, in.Ct, in.co, in.H, Hn, sstr, 1);
        bnstats(stream, tA, mean, var, C, HWn);
        bnapply(stream, tA, out, mean, var, nullptr, nullptr, ws, wb + 0, C, HWn, OC4, no);
        // op1: avg pool -> BN -> acc
        pool(stream, in.p, tA, C, in.Ct, in.co, in.H, Hn, sstr, 0);
        bnstats(stream, tA, mean, var, C, HWn);
        bnapply(stream, tA, out, mean, var, nullptr, nullptr, ws, wb + 1, C, HWn, OC4, no);
        // op2: skip (identity or factorized reduce)
        if (sstr == 1) {
          sacc(stream, in.p, out, ws, wb + 2, C, in.Ct, in.co, HWn, OC4, no);
        } else {
          int h = C / 2;
          gemm(stream, in.p, m.sk1, tA, C, h,     in.Ct, in.co, in.H, Hn, 2, 0, 1, C, 0);
          gemm(stream, in.p, m.sk2, tA, C, C - h, in.Ct, in.co, in.H, Hn, 2, 1, 1, C, h);
          bnstats(stream, tA, mean, var, C, HWn);
          bnapply(stream, tA, out, mean, var, nullptr, nullptr, ws, wb + 2, C, HWn, OC4, no);
        }
        // op3/op4: separable conv k=3 / k=5
        for (int sk = 0; sk < 2; ++sk) {
          const SepP& sp = sk ? m.s5 : m.s3;
          int K = sk ? 5 : 3, pad = K / 2;
          dwc(stream, in.p, sp.dw1, tA, C, in.Ct, in.co, in.H, Hn, K, sstr, pad, 1);
          gemm(stream, tA, sp.pw1, tB, C, C, C, 0, Hn, Hn, 1, 0, 0, C, 0);
          bnstats(stream, tB, mean, var, C, HWn);
          bnapply(stream, tB, tB, mean, var, nullptr, nullptr, nullptr, 0, C, HWn, C, 0);
          dwc(stream, tB, sp.dw2, tA, C, C, 0, Hn, Hn, K, 1, pad, 1);
          gemm(stream, tA, sp.pw2, tB, C, C, C, 0, Hn, Hn, 1, 0, 0, C, 0);
          bnstats(stream, tB, mean, var, C, HWn);
          bnapply(stream, tB, out, mean, var, nullptr, nullptr, ws, wb + 3 + sk, C, HWn, OC4, no);
        }
        // op5/op6: dilated conv k=3 / k=5 (dil=2, pad=k-1)
        for (int dk = 0; dk < 2; ++dk) {
          const DilP& dp = dk ? m.d5 : m.d3;
          int K = dk ? 5 : 3, pad = K - 1;
          dwc(stream, in.p, dp.dw, tA, C, in.Ct, in.co, in.H, Hn, K, sstr, pad, 2);
          gemm(stream, tA, dp.pw, tB, C, C, C, 0, Hn, Hn, 1, 0, 0, C, 0);
          bnstats(stream, tB, mean, var, C, HWn);
          bnapply(stream, tB, out, mean, var, nullptr, nullptr, ws, wb + 5 + dk, C, HWn, OC4, no);
        }
        // op7: zero — contributes nothing
      }
    }
    pp = pv; pv = out; Hpp = Hp; Hp = Hn;
  }

  // --- head: GAP + linear ---
  {
    int total = 32 * 256;
    k_gap<<<(total + 255) / 256, 256, 0, stream>>>(pv, feat, Hp * Hp, total);
    int t2 = 32 * 10;
    k_linear<<<(t2 + 255) / 256, 256, 0, stream>>>(feat, lin_w, lin_b, (float*)d_out, 256, t2);
  }
}
